// ScaledDotProductAttentionEnriched_20916490731985
// MI455X (gfx1250) — matrix operand order repaired and checked by reading, not compiled, not run on hardware
//
#include <hip/hip_runtime.h>
#include <hip/hip_bf16.h>

typedef __attribute__((ext_vector_type(16))) __bf16        v16bf;
typedef __attribute__((ext_vector_type(8)))  float         v8f;
typedef __attribute__((ext_vector_type(8)))  unsigned int  v8u;

__device__ __forceinline__ unsigned short f2bf(float f) {
    unsigned int u = __float_as_uint(f);
    u += 0x7FFFu + ((u >> 16) & 1u);      // round-to-nearest-even
    return (unsigned short)(u >> 16);
}

__device__ __forceinline__ float hmax16(float x) {
    x = fmaxf(x, __shfl_xor(x, 1, 16));
    x = fmaxf(x, __shfl_xor(x, 2, 16));
    x = fmaxf(x, __shfl_xor(x, 4, 16));
    x = fmaxf(x, __shfl_xor(x, 8, 16));
    return x;
}
__device__ __forceinline__ float hsum16(float x) {
    x += __shfl_xor(x, 1, 16);
    x += __shfl_xor(x, 2, 16);
    x += __shfl_xor(x, 4, 16);
    x += __shfl_xor(x, 8, 16);
    return x;
}

// BH=64, S=1024, DK=64, FG=FL=32 -> enriched D=128. Block: 8 waves x 16 rows = 128 queries.
__global__ __launch_bounds__(256, 1)
void fa_enriched_kernel(const float* __restrict__ q,  const float* __restrict__ k,
                        const float* __restrict__ v,  const float* __restrict__ lf,
                        const float* __restrict__ gf, const int*  __restrict__ mask,
                        float* __restrict__ out)
{
    constexpr int S = 1024, DK = 64, FG = 32, FL = 32, D = 128, BM = 128, BN = 32;
    // scores are consumed only as exp(s - max); run softmax in log2 domain:
    // scale = (1/sqrt(64)) * log2(e); masked sentinel = -1e9 * log2(e)
    constexpr float SCALE  = 0.125f * 1.4426950408889634f;
    constexpr float MASKED = -1.4426950408889634e9f;

    const int qb   = blockIdx.x;          // query tile (of BM)
    const int bh   = blockIdx.y;          // batch*head
    const int tid  = threadIdx.x;
    const int lane = tid & 31;
    const int wave = tid >> 5;
    const int half = lane >> 4;           // 0: lanes 0-15, 1: lanes 16-31
    const int ln16 = lane & 15;

    __shared__ unsigned short sPQ[BM][D];       // 32 KB enriched Q (bf16)
    __shared__ unsigned short sPK[BN][D];       //  8 KB enriched K chunk
    __shared__ unsigned short sVT[DK][BN];      //  4 KB V chunk, transposed
    __shared__ unsigned short sPS[8][16][BN];   //  8 KB per-wave P scratch

    const float* qp = q  + (size_t)bh * S * DK;
    const float* kp = k  + (size_t)bh * S * DK;
    const float* vp = v  + (size_t)bh * S * DK;
    const float* gp = gf + (size_t)bh * S * FG;
    const float* lp = lf + (size_t)bh * S * FL;
    const int*   mp = mask + (size_t)bh * S;

    // ---- stage enriched Q tile (fp32 -> bf16) ----
    for (int i = tid; i < BM * D; i += 256) {
        int m = i >> 7, d = i & 127;
        int row = qb * BM + m;
        float val;
        if (d < DK)            val = qp[row * DK + d];
        else if (d < DK + FG)  val = gp[row * FG + (d - DK)];
        else                   val = lp[row * FL + (d - DK - FG)];
        sPQ[m][d] = f2bf(val);
    }
    __syncthreads();

    // ---- persistent A fragments for this wave's 16 query rows (4 chunks of K=32) ----
    v8u aq[4];
    {
        const int m = wave * 16 + ln16;
        #pragma unroll
        for (int c = 0; c < 4; ++c) {
            #pragma unroll
            for (int r = 0; r < 8; ++r) {
                int kk = c * 32 + half * 16 + 2 * r;
                aq[c][r] = *(const unsigned int*)&sPQ[m][kk];
            }
        }
    }

    v8f oacc[4] = {};                     // 16 x 64 output accumulator (4 C tiles)
    float m_i[8], l_i[8];
    #pragma unroll
    for (int r = 0; r < 8; ++r) { m_i[r] = -3.0e38f; l_i[r] = 0.0f; }

    // ==== flash loop over keys, BN=32 per iteration ====
    for (int kb = 0; kb < S / BN; ++kb) {
        __syncthreads();                  // previous iteration done with sPK/sVT

        if (kb + 1 < S / BN) {            // prefetch next raw K chunk (speculative)
            __builtin_prefetch(kp + (size_t)(kb + 1) * BN * DK + tid * 8, 0, 0);
        }

        // stage enriched K chunk (bf16)
        for (int i = tid; i < BN * D; i += 256) {
            int kk = i >> 7, d = i & 127;
            int key = kb * BN + kk;
            float val;
            if (d < DK)            val = kp[key * DK + d];
            else if (d < DK + FG)  val = gp[key * FG + (d - DK)];
            else                   val = lp[key * FL + (d - DK - FG)];
            sPK[kk][d] = f2bf(val);
        }
        // stage V chunk transposed (bf16): sVT[dim][key]
        for (int i = tid; i < BN * DK; i += 256) {
            int kk = i >> 6, d = i & 63;
            sVT[d][kk] = f2bf(vp[(size_t)(kb * BN + kk) * DK + d]);
        }
        // per-lane mask values for this chunk (independent of LDS; hide latency)
        const int mv0 = mp[kb * BN + ln16];
        const int mv1 = mp[kb * BN + 16 + ln16];
        __syncthreads();

        // ---- QK^T: load ALL B fragments first (one DS clause), then stream WMMAs ----
        v8u bu0[4], bu1[4];
        #pragma unroll
        for (int c = 0; c < 4; ++c) {
            #pragma unroll
            for (int r = 0; r < 8; ++r) {
                bu0[c][r] = *(const unsigned int*)&sPK[ln16]     [c * 32 + half * 16 + 2 * r];
                bu1[c][r] = *(const unsigned int*)&sPK[16 + ln16][c * 32 + half * 16 + 2 * r];
            }
        }
        v8f s0 = {}, s1 = {};
        #pragma unroll
        for (int c = 0; c < 4; ++c)
            s0 = __builtin_amdgcn_wmma_f32_16x16x32_bf16(
                     false, __builtin_bit_cast(v16bf, aq[c]),
                     false, __builtin_bit_cast(v16bf, bu0[c]),
                     (short)0, s0, false, false);
        #pragma unroll
        for (int c = 0; c < 4; ++c)
            s1 = __builtin_amdgcn_wmma_f32_16x16x32_bf16(
                     false, __builtin_bit_cast(v16bf, aq[c]),
                     false, __builtin_bit_cast(v16bf, bu1[c]),
                     (short)0, s1, false, false);

        // ---- scale + mask (log2 domain; column key = kb*32 + t*16 + ln16) ----
        #pragma unroll
        for (int r = 0; r < 8; ++r) {
            s0[r] = mv0 ? MASKED : s0[r] * SCALE;
            s1[r] = mv1 ? MASKED : s1[r] * SCALE;
        }

        // ---- online softmax (rows live in 16-lane half-clusters) ----
        #pragma unroll
        for (int r = 0; r < 8; ++r) {
            float tm    = hmax16(fmaxf(s0[r], s1[r]));
            float mnew  = fmaxf(m_i[r], tm);
            float alpha = exp2f(m_i[r] - mnew);
            m_i[r] = mnew;
            float p0 = exp2f(s0[r] - mnew);
            float p1 = exp2f(s1[r] - mnew);
            s0[r] = p0; s1[r] = p1;
            l_i[r] = l_i[r] * alpha + hsum16(p0 + p1);
            oacc[0][r] *= alpha; oacc[1][r] *= alpha;
            oacc[2][r] *= alpha; oacc[3][r] *= alpha;
        }

        // ---- P: C-layout -> A-layout via per-wave LDS scratch (DS is in-order per wave) ----
        #pragma unroll
        for (int r = 0; r < 8; ++r) {
            sPS[wave][r + 8 * half][ln16]      = f2bf(s0[r]);
            sPS[wave][r + 8 * half][16 + ln16] = f2bf(s1[r]);
        }
        // V B-fragments are independent of the P round-trip: issue them in the same clause
        v8u bv[4];
        #pragma unroll
        for (int t2 = 0; t2 < 4; ++t2)
            #pragma unroll
            for (int r = 0; r < 8; ++r)
                bv[t2][r] = *(const unsigned int*)&sVT[t2 * 16 + ln16][half * 16 + 2 * r];
        v8u pa;
        #pragma unroll
        for (int r = 0; r < 8; ++r) {
            int kk = half * 16 + 2 * r;
            pa[r] = *(const unsigned int*)&sPS[wave][ln16][kk];
        }

        // ---- PV: out[16x64] += P[16x32] @ V[32x64], 4 dim tiles ----
        #pragma unroll
        for (int t2 = 0; t2 < 4; ++t2)
            oacc[t2] = __builtin_amdgcn_wmma_f32_16x16x32_bf16(
                           false, __builtin_bit_cast(v16bf, pa),
                           false, __builtin_bit_cast(v16bf, bv[t2]),
                           (short)0, oacc[t2], false, false);
    }

    // ---- epilogue: normalize (8 reciprocals, 32 multiplies) and store fp32 ----
    float* op = out + (size_t)bh * S * DK;
    float inv[8];
    #pragma unroll
    for (int r = 0; r < 8; ++r) inv[r] = 1.0f / l_i[r];
    #pragma unroll
    for (int t2 = 0; t2 < 4; ++t2) {
        #pragma unroll
        for (int r = 0; r < 8; ++r) {
            int m = qb * BM + wave * 16 + r + 8 * half;
            int d = t2 * 16 + ln16;
            op[(size_t)m * DK + d] = oacc[t2][r] * inv[r];
        }
    }
}

extern "C" void kernel_launch(void* const* d_in, const int* in_sizes, int n_in,
                              void* d_out, int out_size, void* d_ws, size_t ws_size,
                              hipStream_t stream) {
    (void)in_sizes; (void)n_in; (void)d_ws; (void)ws_size; (void)out_size;
    const float* q  = (const float*)d_in[0];
    const float* k  = (const float*)d_in[1];
    const float* v  = (const float*)d_in[2];
    const float* lf = (const float*)d_in[3];
    const float* gf = (const float*)d_in[4];
    const int*   mk = (const int*)  d_in[5];
    float* out = (float*)d_out;

    dim3 grid(1024 / 128, 64);   // (S/BM, BH)
    dim3 block(256);
    fa_enriched_kernel<<<grid, block, 0, stream>>>(q, k, v, lf, gf, mk, out);
}
